// Net_81552839016511
// MI455X (gfx1250) — matrix operand order repaired and checked by reading, not compiled, not run on hardware
//
#include <hip/hip_runtime.h>

typedef __attribute__((ext_vector_type(16))) _Float16 v16h;
typedef __attribute__((ext_vector_type(8)))  _Float16 v8h;
typedef __attribute__((ext_vector_type(8)))  float    v8f;

#define HDIM 128
#define LDS_PAD 8   // halves; breaks 256B bank stride (64 banks x 4B)

// ---------------- degree / normalization ----------------

__global__ void deg_init_kernel(float* deg, int n) {
    int i = blockIdx.x * blockDim.x + threadIdx.x;
    if (i < n) deg[i] = 1.0f;   // self-loop included
}

__global__ void deg_count_kernel(const int* __restrict__ dst, float* deg, int E) {
    int e = blockIdx.x * blockDim.x + threadIdx.x;
    if (e < E) atomicAdd(&deg[dst[e]], 1.0f);
}

__global__ void dinv_kernel(float* deg, int n) {
    int i = blockIdx.x * blockDim.x + threadIdx.x;
    if (i < n) deg[i] = rsqrtf(deg[i]);
}

// ---------------- f32 -> f16 activation pre-pass (zero-padded to Kpad) ----------------

__global__ void cvt_f16_kernel(const float* __restrict__ src, _Float16* __restrict__ dst,
                               int M, int K, int Kpad)
{
    long i = (long)blockIdx.x * blockDim.x + threadIdx.x;
    if (i < (long)M * Kpad) {
        int m = (int)(i / Kpad), k = (int)(i % Kpad);
        dst[i] = (_Float16)((k < K) ? src[(long)m * K + k] : 0.0f);
    }
}

// ---------------- WMMA GEMM:  C[M,Nout] = A16[M,Kpad] @ W[K,Nout] (+bias)(+relu) -------
// blockDim = 256 (8 waves). Each wave: one 16-row band x 128 cols = 8 accumulator
// tiles, A fragment reused across 8 v_wmma ops. W staged once per block into LDS,
// f16, TRANSPOSED into B-fragment order: sWt[n*ldn + k]. B fragment = one 32B read.

__global__ __launch_bounds__(256)
void wmma_gemm_kernel(const _Float16* __restrict__ A, const float* __restrict__ W,
                      const float* __restrict__ bias, float* __restrict__ C,
                      int M, int K, int Kpad, int Nout, int do_bias, int do_relu)
{
    __shared__ _Float16 sWt[128 * (128 + LDS_PAD)];   // 34 KB of 320 KB WGP LDS

    const int tid  = threadIdx.x;
    const int lane = tid & 31;
    const int wave = tid >> 5;
    const int ldn  = Kpad + LDS_PAD;

    // cooperative stage of W -> LDS, transposed + f16, zero-padded
    for (int idx = tid; idx < Kpad * 128; idx += 256) {
        int k = idx >> 7, n = idx & 127;
        float v = (k < K && n < Nout) ? W[(long)k * Nout + n] : 0.0f;
        sWt[n * ldn + k] = (_Float16)v;
    }
    __syncthreads();

    const int mTile = blockIdx.x * 8 + wave;
    if (mTile * 16 >= M) return;

    v8f acc[8] = {};

    const int mRow = mTile * 16 + (lane & 15);
    const int kSel = lane >> 4;                        // lane half selects K sub-block
    const _Float16* arow = A + (long)mRow * Kpad;
    const int nChunks = Kpad >> 5;
    const int nCol = lane & 15;

    for (int kc = 0; kc < nChunks; ++kc) {
        // ---- A fragment: two aligned 16B vector loads (contiguous k = 16*half + s order, same as B) ----
        const int ka = kc * 32 + kSel * 16;
        v8h lo = *(const v8h*)(arow + ka);
        v8h hi = *(const v8h*)(arow + ka + 8);
        v16h a;
        #pragma unroll
        for (int j = 0; j < 8; ++j) { a[j] = lo[j]; a[8 + j] = hi[j]; }

        // ---- 8 B fragments: one aligned 32B LDS read each; A reused across WMMAs ----
        const int kb = kc * 32 + kSel * 16;
        #pragma unroll
        for (int nt = 0; nt < 8; ++nt) {
            const int n = nt * 16 + nCol;
            v16h b = *(const v16h*)(&sWt[n * ldn + kb]);
            acc[nt] = __builtin_amdgcn_wmma_f32_16x16x32_f16(
                false, a, false, b, (short)0, acc[nt], false, false);
        }
    }

    // ---- store C (16x16 f32 tile layout: lane half selects M 0-7 / 8-15) ----
    #pragma unroll
    for (int nt = 0; nt < 8; ++nt) {
        const int n = nt * 16 + nCol;
        if (n < Nout) {
            float bval = do_bias ? bias[n] : 0.0f;
            #pragma unroll
            for (int r = 0; r < 8; ++r) {
                int m = mTile * 16 + kSel * 8 + r;
                float v = acc[nt][r] + bval;
                if (do_relu) v = fmaxf(v, 0.0f);
                C[(long)m * Nout + n] = v;
            }
        }
    }
}

// ---------------- aggregation ----------------

// out[n,f] = h[n,f] * dinv[n]^2 + bias[f]   (self-loop term + bias)
__global__ void agg_init_kernel(const float* __restrict__ h, const float* __restrict__ dinv,
                                const float* __restrict__ bias, float* __restrict__ out, int n)
{
    int i = blockIdx.x * blockDim.x + threadIdx.x;
    if (i < n * HDIM) {
        int node = i >> 7, f = i & 127;
        float di = dinv[node];
        out[i] = h[i] * di * di + bias[f];
    }
}

// one wave per edge; lane handles a contiguous float4 of features (L2-resident gather)
__global__ void agg_edge_kernel(const int* __restrict__ src, const int* __restrict__ dst,
                                const float* __restrict__ dinv, const float* __restrict__ h,
                                float* __restrict__ out, int E)
{
    long tid = (long)blockIdx.x * blockDim.x + threadIdx.x;
    int e = (int)(tid >> 5);
    if (e >= E) return;
    int lane = (int)(tid & 31);
    int s = src[e], d = dst[e];
    float norm = dinv[s] * dinv[d];
    const float4 hv = ((const float4*)(h + (long)s * HDIM))[lane];
    float* op = out + (long)d * HDIM + lane * 4;
    atomicAdd(op + 0, hv.x * norm);
    atomicAdd(op + 1, hv.y * norm);
    atomicAdd(op + 2, hv.z * norm);
    atomicAdd(op + 3, hv.w * norm);
}

__global__ void relu_kernel(float* x, long n) {
    long i = (long)blockIdx.x * blockDim.x + threadIdx.x;
    if (i < n) x[i] = fmaxf(x[i], 0.0f);
}

// ---------------- driver ----------------

extern "C" void kernel_launch(void* const* d_in, const int* in_sizes, int n_in,
                              void* d_out, int out_size, void* d_ws, size_t ws_size,
                              hipStream_t stream)
{
    const float* x    = (const float*)d_in[0];
    const int*   edge = (const int*)d_in[1];
    const float* W1   = (const float*)d_in[3];
    const float* b1   = (const float*)d_in[4];
    const float* W2   = (const float*)d_in[5];
    const float* b2   = (const float*)d_in[6];
    const float* W3   = (const float*)d_in[7];
    const float* b3   = (const float*)d_in[8];
    const float* fcW1 = (const float*)d_in[9];
    const float* fcb1 = (const float*)d_in[10];
    const float* fcW2 = (const float*)d_in[11];
    const float* fcb2 = (const float*)d_in[12];
    const float* fcW3 = (const float*)d_in[13];
    const float* fcb3 = (const float*)d_in[14];
    float* out = (float*)d_out;

    const int E    = in_sizes[1] / 2;
    const int K1   = in_sizes[3] / HDIM;              // F_IN = 50
    const int N    = in_sizes[0] / K1;                // 50000
    const int COUT = in_sizes[13] / HDIM;             // 121
    const int Kp1  = ((K1 + 31) / 32) * 32;           // 64

    const int* srcIdx = edge;
    const int* dstIdx = edge + E;

    // workspace layout: dinv[N] | bufA[N*128 f32] | bufB[N*128 f32] | h16[N*128 f16]
    char* ws = (char*)d_ws;
    float* dinv = (float*)ws;
    size_t off = ((size_t)N * 4 + 255) & ~(size_t)255;
    float* bufA = (float*)(ws + off);
    off += (size_t)N * HDIM * 4;
    float* bufB = (float*)(ws + off);
    off += (size_t)N * HDIM * 4;
    _Float16* h16 = (_Float16*)(ws + off);

    const int nBlkN   = (N + 255) / 256;
    const int nBlkE   = (E + 255) / 256;
    const int mBlocks = (((N + 15) / 16) + 7) / 8;
    const long NH     = (long)N * HDIM;
    const int ewBlk   = (int)((NH + 255) / 256);
    const int aggBlk  = (int)(((long)E * 32 + 255) / 256);
    const int cvBlk1  = (int)(((long)N * Kp1 + 255) / 256);

    // degrees -> dinv (in place)
    deg_init_kernel<<<nBlkN, 256, 0, stream>>>(dinv, N);
    deg_count_kernel<<<nBlkE, 256, 0, stream>>>(dstIdx, dinv, E);
    dinv_kernel<<<nBlkN, 256, 0, stream>>>(dinv, N);

    // GCN layer 1
    cvt_f16_kernel<<<cvBlk1, 256, 0, stream>>>(x, h16, N, K1, Kp1);
    wmma_gemm_kernel<<<mBlocks, 256, 0, stream>>>(h16, W1, nullptr, bufA, N, K1, Kp1, HDIM, 0, 0);
    agg_init_kernel<<<ewBlk, 256, 0, stream>>>(bufA, dinv, b1, bufB, N);
    agg_edge_kernel<<<aggBlk, 256, 0, stream>>>(srcIdx, dstIdx, dinv, bufA, bufB, E);
    relu_kernel<<<ewBlk, 256, 0, stream>>>(bufB, NH);

    // GCN layer 2
    cvt_f16_kernel<<<ewBlk, 256, 0, stream>>>(bufB, h16, N, HDIM, HDIM);
    wmma_gemm_kernel<<<mBlocks, 256, 0, stream>>>(h16, W2, nullptr, bufA, N, HDIM, HDIM, HDIM, 0, 0);
    agg_init_kernel<<<ewBlk, 256, 0, stream>>>(bufA, dinv, b2, bufB, N);
    agg_edge_kernel<<<aggBlk, 256, 0, stream>>>(srcIdx, dstIdx, dinv, bufA, bufB, E);
    relu_kernel<<<ewBlk, 256, 0, stream>>>(bufB, NH);

    // GCN layer 3
    cvt_f16_kernel<<<ewBlk, 256, 0, stream>>>(bufB, h16, N, HDIM, HDIM);
    wmma_gemm_kernel<<<mBlocks, 256, 0, stream>>>(h16, W3, nullptr, bufA, N, HDIM, HDIM, HDIM, 0, 0);
    agg_init_kernel<<<ewBlk, 256, 0, stream>>>(bufA, dinv, b3, bufB, N);
    agg_edge_kernel<<<aggBlk, 256, 0, stream>>>(srcIdx, dstIdx, dinv, bufA, bufB, E);
    relu_kernel<<<ewBlk, 256, 0, stream>>>(bufB, NH);

    // FC head
    cvt_f16_kernel<<<ewBlk, 256, 0, stream>>>(bufB, h16, N, HDIM, HDIM);
    wmma_gemm_kernel<<<mBlocks, 256, 0, stream>>>(h16, fcW1, fcb1, bufA, N, HDIM, HDIM, HDIM, 1, 1);
    cvt_f16_kernel<<<ewBlk, 256, 0, stream>>>(bufA, h16, N, HDIM, HDIM);
    wmma_gemm_kernel<<<mBlocks, 256, 0, stream>>>(h16, fcW2, fcb2, bufB, N, HDIM, HDIM, HDIM, 1, 1);
    cvt_f16_kernel<<<ewBlk, 256, 0, stream>>>(bufB, h16, N, HDIM, HDIM);
    wmma_gemm_kernel<<<mBlocks, 256, 0, stream>>>(h16, fcW3, fcb3, out, N, HDIM, HDIM, COUT, 1, 0);
}
